// RoutedAttention_53979148976795
// MI455X (gfx1250) — compile-verified, matrix operand order checked by reading, not run on hardware
//
#include <hip/hip_runtime.h>
#include <math.h>

typedef __bf16 bf16_t;
typedef __attribute__((ext_vector_type(8)))  bf16_t v8bf;
typedef __attribute__((ext_vector_type(16))) bf16_t v16bf;
typedef __attribute__((ext_vector_type(8)))  float  v8f;

constexpr int DIMC = 1024;
constexpr int LSEQ = 2048;
constexpr int NH   = 16;
constexpr int DK   = 64;
constexpr int NB   = 2;

static __device__ __forceinline__ v16bf make_frag(const bf16_t* p0, const bf16_t* p1) {
  v8bf a = *(const v8bf*)p0;
  v8bf b = *(const v8bf*)p1;
  v16bf r;
#pragma unroll
  for (int i = 0; i < 8; ++i) { r[i] = a[i]; r[i + 8] = b[i]; }
  return r;
}

static __device__ __forceinline__ v8bf cvt8(float4 f0, float4 f1) {
  v8bf p;
  p[0] = (bf16_t)f0.x; p[1] = (bf16_t)f0.y; p[2] = (bf16_t)f0.z; p[3] = (bf16_t)f0.w;
  p[4] = (bf16_t)f1.x; p[5] = (bf16_t)f1.y; p[6] = (bf16_t)f1.z; p[7] = (bf16_t)f1.w;
  return p;
}

// Y = X (M x 1024, f32 or bf16) @ W^T (W: 1024 x 1024 f32)
// MODE 0: store bf16 permuted [b,h,l,dk]             (K projection)
// MODE 1: store bf16 permuted [b,h,l,dk], * 0.125    (Q projection, pre-scaled)
// MODE 2: store bf16 transposed [b,h,dk,l]           (V projection)
// MODE 3: X is bf16, store f32 row-major             (output projection)
template <int MODE>
__global__ __launch_bounds__(256) void gemm_wmma(
    const float* __restrict__ Xf, const bf16_t* __restrict__ Xbf,
    const float* __restrict__ W,
    bf16_t* __restrict__ outB, float* __restrict__ outF) {
  __shared__ bf16_t As[128][32];
  __shared__ bf16_t Bs[128][32];

  const int tid  = threadIdx.x;
  const int lane = tid & 31;
  const int wave = tid >> 5;       // 0..7
  const int wm   = wave >> 1;      // 0..3
  const int wn   = wave & 1;       // 0..1
  const int m0   = blockIdx.x * 128;
  const int n0   = blockIdx.y * 128;

  const int srow = tid >> 1;              // staging row 0..127
  const int scol = (tid & 1) * 16;        // 0 or 16

  const int rr   = lane & 15;
  const int rsel = lane >> 4;             // 0/1
  const int kh   = rsel * 8;

  const v8f vz = {0.f, 0.f, 0.f, 0.f, 0.f, 0.f, 0.f, 0.f};
  v8f acc[2][4];
#pragma unroll
  for (int i = 0; i < 2; ++i)
#pragma unroll
    for (int j = 0; j < 4; ++j) acc[i][j] = vz;

  for (int k0 = 0; k0 < DIMC; k0 += 32) {
    // ---- stage A tile ----
    if (MODE == 3) {
      const bf16_t* src = Xbf + (size_t)(m0 + srow) * DIMC + k0 + scol;
      *(v8bf*)&As[srow][scol]     = *(const v8bf*)(src);
      *(v8bf*)&As[srow][scol + 8] = *(const v8bf*)(src + 8);
    } else {
      const float* src = Xf + (size_t)(m0 + srow) * DIMC + k0 + scol;
      float4 f0 = ((const float4*)src)[0];
      float4 f1 = ((const float4*)src)[1];
      float4 f2 = ((const float4*)src)[2];
      float4 f3 = ((const float4*)src)[3];
      *(v8bf*)&As[srow][scol]     = cvt8(f0, f1);
      *(v8bf*)&As[srow][scol + 8] = cvt8(f2, f3);
    }
    // ---- stage B tile (rows of W, k-major) ----
    {
      const float* src = W + (size_t)(n0 + srow) * DIMC + k0 + scol;
      float4 f0 = ((const float4*)src)[0];
      float4 f1 = ((const float4*)src)[1];
      float4 f2 = ((const float4*)src)[2];
      float4 f3 = ((const float4*)src)[3];
      *(v8bf*)&Bs[srow][scol]     = cvt8(f0, f1);
      *(v8bf*)&Bs[srow][scol + 8] = cvt8(f2, f3);
    }
    __syncthreads();

    const int am = wm * 32;
    const int bn = wn * 64;
    v16bf afr[2], bfr[4];
#pragma unroll
    for (int i = 0; i < 2; ++i)
      afr[i] = make_frag(&As[am + i * 16 + rr][kh], &As[am + i * 16 + rr][kh + 16]);
#pragma unroll
    for (int j = 0; j < 4; ++j)
      bfr[j] = make_frag(&Bs[bn + j * 16 + rr][kh], &Bs[bn + j * 16 + rr][kh + 16]);
#pragma unroll
    for (int i = 0; i < 2; ++i)
#pragma unroll
      for (int j = 0; j < 4; ++j)
        acc[i][j] = __builtin_amdgcn_wmma_f32_16x16x32_bf16(
            false, afr[i], false, bfr[j], (short)0, acc[i][j], false, false);
    __syncthreads();
  }

  // ---- epilogue ----
  const int rbase = rsel * 8;
#pragma unroll
  for (int i = 0; i < 2; ++i) {
#pragma unroll
    for (int j = 0; j < 4; ++j) {
      const int N = n0 + wn * 64 + j * 16 + rr;
#pragma unroll
      for (int r = 0; r < 8; ++r) {
        const int M = m0 + wm * 32 + i * 16 + rbase + r;
        float val = acc[i][j][r];
        if (MODE == 1) val *= 0.125f;   // 1/sqrt(dk) folded into Q
        if (MODE == 3) {
          outF[(size_t)M * DIMC + N] = val;
        } else {
          const int b  = M >> 11;       // / 2048
          const int l  = M & (LSEQ - 1);
          const int dk = N >> 4;
          const int h  = N & 15;
          if (MODE == 2)
            outB[(((size_t)(b * NH + h) * DK + dk) * LSEQ) + l] = (bf16_t)val;
          else
            outB[(((size_t)(b * NH + h) * LSEQ + l) * DK) + dk] = (bf16_t)val;
        }
      }
    }
  }
}

// Flash-style attention per (b, h): Qh[b,h,l,dk] (pre-scaled), Kh[b,h,l,dk], Vt[b,h,dk,l]
// out: attn[b, l, dk*16 + h] bf16. 64-key tiles per iteration.
__global__ __launch_bounds__(256) void attn_wmma(
    const bf16_t* __restrict__ Qh, const bf16_t* __restrict__ Kh,
    const bf16_t* __restrict__ Vt, const unsigned char* __restrict__ mask,
    bf16_t* __restrict__ attnOut) {
  __shared__ bf16_t Ks[64][64];        // [key][dk]
  __shared__ bf16_t Vs[64][64];        // [dk][key]
  __shared__ bf16_t Ps[8][16][64];     // per-wave P staging

  const int tid  = threadIdx.x;
  const int lane = tid & 31;
  const int wave = tid >> 5;
  const int b    = blockIdx.z;
  const int h    = blockIdx.y;
  const int q0   = blockIdx.x * 128 + wave * 16;

  const bf16_t* Qbase = Qh + (size_t)(b * NH + h) * LSEQ * DK;
  const bf16_t* Kbase = Kh + (size_t)(b * NH + h) * LSEQ * DK;
  const bf16_t* Vbase = Vt + (size_t)(b * NH + h) * DK * LSEQ;
  const unsigned char* Mbase = mask + (size_t)b * LSEQ * LSEQ;

  const int rr   = lane & 15;
  const int rsel = lane >> 4;
  const int kh   = rsel * 8;

  // staging coords: thread copies 16 bf16 (2 x b128) per matrix
  const int trow = tid >> 2;           // 0..63
  const int tcol = (tid & 3) * 16;     // 0,16,32,48

  // Q fragments: 2 k-steps cover dk = 64 (held in registers all kernel)
  v16bf qf[2];
#pragma unroll
  for (int s = 0; s < 2; ++s) {
    const bf16_t* qp = Qbase + (size_t)(q0 + rr) * DK + s * 32 + kh;
    qf[s] = make_frag(qp, qp + 16);
  }

  const v8f vz = {0.f, 0.f, 0.f, 0.f, 0.f, 0.f, 0.f, 0.f};
  float mrow[8], lrow[8];
#pragma unroll
  for (int r = 0; r < 8; ++r) { mrow[r] = -3.0e38f; lrow[r] = 0.f; }
  v8f oacc[4];
#pragma unroll
  for (int j = 0; j < 4; ++j) oacc[j] = vz;

  for (int o0 = 0; o0 < LSEQ; o0 += 64) {
    // ---- stage K tile (64 keys x 64 dk) and V tile (64 dk x 64 keys) ----
    {
      const bf16_t* kp = Kbase + (size_t)(o0 + trow) * DK + tcol;
      *(v8bf*)&Ks[trow][tcol]     = *(const v8bf*)(kp);
      *(v8bf*)&Ks[trow][tcol + 8] = *(const v8bf*)(kp + 8);
      const bf16_t* vp = Vbase + (size_t)trow * LSEQ + o0 + tcol;
      *(v8bf*)&Vs[trow][tcol]     = *(const v8bf*)(vp);
      *(v8bf*)&Vs[trow][tcol + 8] = *(const v8bf*)(vp + 8);
    }
    __syncthreads();

    // ---- S = (Q/8) K^T for four 16-key tiles ----
    float sv[4][8];
#pragma unroll
    for (int n = 0; n < 4; ++n) {
      v8f c = vz;
#pragma unroll
      for (int s = 0; s < 2; ++s) {
        v16bf kf = make_frag(&Ks[n * 16 + rr][s * 32 + kh],
                             &Ks[n * 16 + rr][s * 32 + kh + 16]);
        c = __builtin_amdgcn_wmma_f32_16x16x32_bf16(
            false, qf[s], false, kf, (short)0, c, false, false);
      }
      const int ocol = o0 + n * 16 + rr;
#pragma unroll
      for (int r = 0; r < 8; ++r) {
        float x = c[r];
        const int qi = q0 + rsel * 8 + r;
        if (Mbase[(size_t)qi * LSEQ + ocol]) x = -1.0e10f;
        sv[n][r] = x;
      }
    }

    // ---- online softmax over the 64 keys ----
#pragma unroll
    for (int r = 0; r < 8; ++r) {
      float mx = fmaxf(fmaxf(sv[0][r], sv[1][r]), fmaxf(sv[2][r], sv[3][r]));
#pragma unroll
      for (int d = 1; d < 16; d <<= 1) mx = fmaxf(mx, __shfl_xor(mx, d, 32));
      const float mnew  = fmaxf(mrow[r], mx);
      const float alpha = __expf(mrow[r] - mnew);
      float ps = 0.f;
#pragma unroll
      for (int n = 0; n < 4; ++n) {
        sv[n][r] = __expf(sv[n][r] - mnew);
        ps += sv[n][r];
      }
#pragma unroll
      for (int d = 1; d < 16; d <<= 1) ps += __shfl_xor(ps, d, 32);
      lrow[r] = lrow[r] * alpha + ps;
      mrow[r] = mnew;
#pragma unroll
      for (int j = 0; j < 4; ++j) oacc[j][r] *= alpha;
    }

    // ---- C-layout P -> A-layout fragments via wave-private LDS slice ----
#pragma unroll
    for (int n = 0; n < 4; ++n)
#pragma unroll
      for (int r = 0; r < 8; ++r)
        Ps[wave][rsel * 8 + r][n * 16 + rr] = (bf16_t)sv[n][r];

    v16bf pf[2];
#pragma unroll
    for (int s = 0; s < 2; ++s)
      pf[s] = make_frag(&Ps[wave][rr][s * 32 + kh], &Ps[wave][rr][s * 32 + kh + 16]);

    // ---- O += P V ----
#pragma unroll
    for (int j = 0; j < 4; ++j) {
#pragma unroll
      for (int s = 0; s < 2; ++s) {
        v16bf vf = make_frag(&Vs[j * 16 + rr][s * 32 + kh],
                             &Vs[j * 16 + rr][s * 32 + kh + 16]);
        oacc[j] = __builtin_amdgcn_wmma_f32_16x16x32_bf16(
            false, pf[s], false, vf, (short)0, oacc[j], false, false);
      }
    }
    __syncthreads();
  }

  // ---- normalize and store: attn[b, qi, dk*16 + h] ----
#pragma unroll
  for (int r = 0; r < 8; ++r) {
    const float inv = 1.f / lrow[r];
    const int qi = q0 + rsel * 8 + r;
#pragma unroll
    for (int j = 0; j < 4; ++j) {
      const int dk = j * 16 + rr;
      attnOut[(size_t)(b * LSEQ + qi) * DIMC + dk * 16 + h] = (bf16_t)(oacc[j][r] * inv);
    }
  }
}

extern "C" void kernel_launch(void* const* d_in, const int* in_sizes, int n_in,
                              void* d_out, int out_size, void* d_ws, size_t ws_size,
                              hipStream_t stream) {
  const float* q  = (const float*)d_in[0];
  const float* k  = (const float*)d_in[1];
  const float* v  = (const float*)d_in[2];
  const unsigned char* mask = (const unsigned char*)d_in[3];
  const float* Wq = (const float*)d_in[4];
  const float* Wk = (const float*)d_in[5];
  const float* Wv = (const float*)d_in[6];
  const float* Wo = (const float*)d_in[7];
  float* out = (float*)d_out;

  const size_t elems = (size_t)NB * NH * LSEQ * DK;  // 4M bf16 each
  bf16_t* Qh   = (bf16_t*)d_ws;
  bf16_t* Kh   = Qh + elems;
  bf16_t* Vt   = Kh + elems;
  bf16_t* attn = Vt + elems;

  const dim3 gG(32, 8), bG(256);
  gemm_wmma<1><<<gG, bG, 0, stream>>>(q, nullptr, Wq, Qh, nullptr);   // Q, pre-scaled
  gemm_wmma<0><<<gG, bG, 0, stream>>>(k, nullptr, Wk, Kh, nullptr);   // K
  gemm_wmma<2><<<gG, bG, 0, stream>>>(v, nullptr, Wv, Vt, nullptr);   // V (transposed)
  attn_wmma<<<dim3(LSEQ / 128, NH, NB), 256, 0, stream>>>(Qh, Kh, Vt, mask, attn);
  gemm_wmma<3><<<gG, bG, 0, stream>>>(nullptr, attn, Wo, nullptr, out);
}
